// LinearAttnFFN_7885559955666
// MI455X (gfx1250) — matrix operand order repaired and checked by reading, not compiled, not run on hardware
//
#include <hip/hip_runtime.h>
#include <hip/hip_bf16.h>

typedef __attribute__((ext_vector_type(16))) _Float16 v16h;
typedef __attribute__((ext_vector_type(8)))  _Float16 v8h;
typedef __attribute__((ext_vector_type(8)))  float    v8f;

#define CDIM 256
#define PDIM 256
#define NDIM 256
#define FDIM 512
#define NS   264   // padded LDS row stride (halves) for 256-wide transposed buffers
#define HS   520   // padded LDS row stride (halves) for 512-wide h buffer

// f16 weight layout in d_ws (halves):
#define W16_K   256            // qkv rows 1..256
#define W16_V   (257*256)      // qkv rows 257..512
#define W16_OUT (513*256)
#define W16_F1  (W16_OUT + 256*256)
#define W16_F2  (W16_F1 + 512*256)
#define W16_TOTAL (W16_F2 + 256*512)

#define SMEM_BYTES (CDIM*NS*2*2 + (NDIM + NDIM + CDIM + 16)*4)

union V16H { v16h v; v8h h[2]; };
union PK2  { _Float16 h[2]; unsigned u; };

__device__ __forceinline__ float wave_red_add(float v) {
  #pragma unroll
  for (int s = 16; s >= 1; s >>= 1) v += __shfl_xor(v, s, 32);
  return v;
}

// ---------------------------------------------------------------------------
// GEMM core with register-resident A: D(O x N) = W(O x Cin) * B(LDS [n][c]).
// Each wave owns whole o-rows (ot = wave, wave+8, ...); the NFRAG A fragments
// for a row load once into VGPRs and are reused across every n-tile, so the
// inner loop is pure ds_load_b128 -> v_wmma_f32_16x16x32_f16. Epilogue is
// invoked per tile with the full accumulator, letting call sites hoist
// o-dependent state at nt==0 and carry register state across n-tiles.
// Operand mapping (A/B k order is contiguous, not the ISA table order):
//   A lane m=lane&15, kh=lane>>4 : halves h -> K = c0 + 16*kh + h
//                                  (contiguous, matches B)
//   B lane: N = n0 + (lane&15), halves h -> K = c0 + 16*kh + h (contiguous)
//   D comp r: row = ot*16 + r + 8*kh, col = nt*16 + (lane&15)
// ---------------------------------------------------------------------------
template <int NFRAG, typename TileEpi>
__device__ __forceinline__ void gemm_areg(const _Float16* __restrict__ W, int ldw,
                                          const _Float16* Bsh, int bstride,
                                          int Otiles, int Ntiles,
                                          int wave, int lane, TileEpi epi) {
  const int m  = lane & 15;
  const int kh = lane >> 4;
  for (int ot = wave; ot < Otiles; ot += 8) {
    const _Float16* wrow = W + (size_t)(ot * 16 + m) * ldw;
    V16H A[NFRAG];
    #pragma unroll
    for (int f = 0; f < NFRAG; ++f) {
      A[f].h[0] = *(const v8h*)(wrow + f * 32 + kh * 16);
      A[f].h[1] = *(const v8h*)(wrow + f * 32 + kh * 16 + 8);
    }
    if (ot + 8 < Otiles)  // warm L0/L2 for this wave's next o-row
      __builtin_prefetch(wrow + (size_t)128 * ldw, 0, 1);
    for (int nt = 0; nt < Ntiles; ++nt) {
      const _Float16* brow = Bsh + (size_t)(nt * 16 + m) * bstride + kh * 16;
      v8f acc = {};
      #pragma unroll
      for (int f = 0; f < NFRAG; ++f) {
        V16H B;
        B.h[0] = *(const v8h*)(brow + f * 32);
        B.h[1] = *(const v8h*)(brow + f * 32 + 8);
        acc = __builtin_amdgcn_wmma_f32_16x16x32_f16(
            false, A[f].v, false, B.v, (short)0, acc, false, false);
      }
      epi(ot, nt, acc);
    }
  }
}

// LayerNorm over the last axis of one (C x N) slice, rows from global f32
// (row stride P*N), writing f16 TRANSPOSED [n][c] into LDS.
// NOTE: gamma/beta have shape (C,) but numpy-broadcast along the LAST axis,
// so they are indexed by n (matches the reference semantics).
__device__ __forceinline__ void layernorm_to_lds(const float* __restrict__ src,
                                                 const float* __restrict__ g,
                                                 const float* __restrict__ bb,
                                                 _Float16* dst, int wave, int lane) {
  for (int c = wave; c < CDIM; c += 8) {
    const float* xr = src + (size_t)c * (PDIM * NDIM);
    float4 a0 = *(const float4*)(xr + lane * 8);
    float4 a1 = *(const float4*)(xr + lane * 8 + 4);
    float e[8] = {a0.x, a0.y, a0.z, a0.w, a1.x, a1.y, a1.z, a1.w};
    float s = 0.f, s2 = 0.f;
    #pragma unroll
    for (int j = 0; j < 8; ++j) { s += e[j]; s2 += e[j] * e[j]; }
    s = wave_red_add(s);  s2 = wave_red_add(s2);
    float mu  = s * (1.0f / NDIM);
    float var = s2 * (1.0f / NDIM) - mu * mu;
    float rs  = rsqrtf(var + 1e-5f);
    #pragma unroll
    for (int j = 0; j < 8; ++j) {
      int n = lane * 8 + j;
      dst[(size_t)n * NS + c] = (_Float16)((e[j] - mu) * rs * g[n] + bb[n]);
    }
  }
}

__global__ void cvt_f32_to_f16(const float* __restrict__ s,
                               _Float16* __restrict__ d, int n) {
  int i = blockIdx.x * blockDim.x + threadIdx.x;
  if (i < n) d[i] = (_Float16)s[i];
}

__global__ __launch_bounds__(256, 1) void linattn_ffn_fused(
    const float* __restrict__ x,
    const float* __restrict__ ln1_g, const float* __restrict__ ln1_b,
    const float* __restrict__ qkv_w, const float* __restrict__ qkv_b,
    const float* __restrict__ out_b,
    const float* __restrict__ ln2_g, const float* __restrict__ ln2_b,
    const float* __restrict__ ffn1_b, const float* __restrict__ ffn2_b,
    const _Float16* __restrict__ w16, float* __restrict__ out) {
  extern __shared__ char smem[];
  _Float16* Y = (_Float16*)smem;          // y / z  (f16, [n][c], stride NS)
  _Float16* K = Y + CDIM * NS;            // attn / h (f16)
  float* qrow   = (float*)(K + CDIM * NS);
  float* scores = qrow + NDIM;
  float* ctx    = scores + NDIM;
  float* red    = ctx + CDIM;             // 16 floats scratch

  const int tid = threadIdx.x, lane = tid & 31, wave = tid >> 5;
  const int lm = lane & 15, lkh = lane >> 4;
  const int b = blockIdx.x / PDIM, p = blockIdx.x % PDIM;
  const float* xb = x   + ((size_t)b * CDIM * PDIM + p) * NDIM; // (b,0,p,0)
  float*       ob = out + ((size_t)b * CDIM * PDIM + p) * NDIM;

  // ---- Phase 1: LN1 -> Y ----
  layernorm_to_lds(xb, ln1_g, ln1_b, Y, wave, lane);
  __syncthreads();

  // ---- Phase 1b: q[n] = qkv_w[0,:] . y[:,n] ----
  {
    float acc = qkv_b[0];
    const _Float16* yr = Y + (size_t)tid * NS;
    for (int c0 = 0; c0 < CDIM; c0 += 8) {
      v8h yv = *(const v8h*)(yr + c0);
      #pragma unroll
      for (int j = 0; j < 8; ++j) acc += qkv_w[c0 + j] * (float)yv[j];
    }
    qrow[tid] = acc;
  }
  __syncthreads();

  // ---- softmax over n ----
  {
    float qv = qrow[tid];
    float mx = qv;
    #pragma unroll
    for (int s = 16; s >= 1; s >>= 1) mx = fmaxf(mx, __shfl_xor(mx, s, 32));
    if (lane == 0) red[wave] = mx;
    __syncthreads();
    if (tid == 0) { float t = red[0]; for (int i = 1; i < 8; ++i) t = fmaxf(t, red[i]); red[8] = t; }
    __syncthreads();
    float e = __expf(qv - red[8]);
    float s = wave_red_add(e);
    if (lane == 0) red[wave] = s;
    __syncthreads();
    if (tid == 0) { float t = 0.f; for (int i = 0; i < 8; ++i) t += red[i]; red[9] = t; }
    __syncthreads();
    scores[tid] = e / red[9];
  }
  __syncthreads();

  // ---- Phase 2: k GEMM; ctx[o] = sum_n k[o,n]*score[n] (+bias, since
  //      softmax weights sum to 1). Each (wave,kh,r) owns o exclusively:
  //      register accumulation across n-tiles, ONE shuffle-reduce per o-row,
  //      plain ds_store (no atomics). ----
  {
    float part[8];
    gemm_areg<8>(w16 + W16_K, CDIM, Y, NS, CDIM / 16, NDIM / 16, wave, lane,
      [&](int ot, int nt, const v8f& acc) {
        if (nt == 0) {
          #pragma unroll
          for (int r = 0; r < 8; ++r) part[r] = 0.f;
        }
        float sc = scores[nt * 16 + lm];
        #pragma unroll
        for (int r = 0; r < 8; ++r) part[r] = fmaf(acc[r], sc, part[r]);
        if (nt == NDIM / 16 - 1) {
          #pragma unroll
          for (int r = 0; r < 8; ++r) {
            float pv = part[r];
            #pragma unroll
            for (int s = 8; s >= 1; s >>= 1) pv += __shfl_xor(pv, s, 32);
            if (lm == 0) {
              int o = ot * 16 + r + 8 * lkh;
              ctx[o] = pv + qkv_b[1 + o];
            }
          }
        }
      });
  }
  __syncthreads();

  // ---- Phase 3: v GEMM; attn = relu(v+b)*ctx -> K (transposed f16) ----
  {
    float vb[8], cx[8];
    gemm_areg<8>(w16 + W16_V, CDIM, Y, NS, CDIM / 16, NDIM / 16, wave, lane,
      [&](int ot, int nt, const v8f& acc) {
        if (nt == 0) {
          #pragma unroll
          for (int r = 0; r < 8; ++r) {
            int o = ot * 16 + r + 8 * lkh;
            vb[r] = qkv_b[257 + o];
            cx[r] = ctx[o];
          }
        }
        _Float16* krow = K + (size_t)(nt * 16 + lm) * NS + ot * 16 + 8 * lkh;
        #pragma unroll
        for (int r = 0; r < 8; r += 2) {
          PK2 pk;
          pk.h[0] = (_Float16)(fmaxf(acc[r]     + vb[r],     0.f) * cx[r]);
          pk.h[1] = (_Float16)(fmaxf(acc[r + 1] + vb[r + 1], 0.f) * cx[r + 1]);
          *(unsigned*)(krow + r) = pk.u;
        }
      });
  }
  __syncthreads();

  // ---- Phase 4: out GEMM + residual -> d_out (holds x2) ----
  {
    float obv[8];
    gemm_areg<8>(w16 + W16_OUT, CDIM, K, NS, CDIM / 16, NDIM / 16, wave, lane,
      [&](int ot, int nt, const v8f& acc) {
        if (nt == 0) {
          #pragma unroll
          for (int r = 0; r < 8; ++r) obv[r] = out_b[ot * 16 + r + 8 * lkh];
        }
        int n = nt * 16 + lm;
        #pragma unroll
        for (int r = 0; r < 8; ++r) {
          size_t idx = (size_t)(ot * 16 + r + 8 * lkh) * (PDIM * NDIM) + n;
          ob[idx] = xb[idx] + acc[r] + obv[r];
        }
      });
  }
  __threadfence();
  __syncthreads();

  // ---- Phase 5: LN2(x2) -> Y ----
  layernorm_to_lds(ob, ln2_g, ln2_b, Y, wave, lane);
  __syncthreads();

  // ---- Phase 6: FFN, two halves of N so h (512 x 128 f16) fits in K ----
  for (int nh = 0; nh < 2; ++nh) {
    {
      float fb[8];
      gemm_areg<8>(w16 + W16_F1, CDIM, Y + (size_t)(nh * 128) * NS, NS,
                   FDIM / 16, 128 / 16, wave, lane,
        [&](int ot, int nt, const v8f& acc) {
          if (nt == 0) {
            #pragma unroll
            for (int r = 0; r < 8; ++r) fb[r] = ffn1_b[ot * 16 + r + 8 * lkh];
          }
          _Float16* hrow = K + (size_t)(nt * 16 + lm) * HS + ot * 16 + 8 * lkh;
          #pragma unroll
          for (int r = 0; r < 8; r += 2) {
            PK2 pk;
            pk.h[0] = (_Float16)fmaxf(acc[r]     + fb[r],     0.f);
            pk.h[1] = (_Float16)fmaxf(acc[r + 1] + fb[r + 1], 0.f);
            *(unsigned*)(hrow + r) = pk.u;
          }
        });
    }
    __syncthreads();
    {
      float fb[8];
      gemm_areg<16>(w16 + W16_F2, FDIM, K, HS, CDIM / 16, 128 / 16, wave, lane,
        [&](int ot, int nt, const v8f& acc) {
          if (nt == 0) {
            #pragma unroll
            for (int r = 0; r < 8; ++r) fb[r] = ffn2_b[ot * 16 + r + 8 * lkh];
          }
          int n = nh * 128 + nt * 16 + lm;
          #pragma unroll
          for (int r = 0; r < 8; ++r) {
            size_t idx = (size_t)(ot * 16 + r + 8 * lkh) * (PDIM * NDIM) + n;
            ob[idx] += acc[r] + fb[r];
          }
        });
    }
    __syncthreads();
  }
}

extern "C" void kernel_launch(void* const* d_in, const int* in_sizes, int n_in,
                              void* d_out, int out_size, void* d_ws, size_t ws_size,
                              hipStream_t stream) {
  const float* x      = (const float*)d_in[0];
  const float* ln1_g  = (const float*)d_in[1];
  const float* ln1_b  = (const float*)d_in[2];
  const float* qkv_w  = (const float*)d_in[3];
  const float* qkv_b  = (const float*)d_in[4];
  const float* out_w  = (const float*)d_in[5];
  const float* out_bv = (const float*)d_in[6];
  const float* ln2_g  = (const float*)d_in[7];
  const float* ln2_b  = (const float*)d_in[8];
  const float* ffn1_w = (const float*)d_in[9];
  const float* ffn1_b = (const float*)d_in[10];
  const float* ffn2_w = (const float*)d_in[11];
  const float* ffn2_b = (const float*)d_in[12];
  _Float16* w16 = (_Float16*)d_ws;   // needs W16_TOTAL*2 = ~0.92 MB
  float* outp = (float*)d_out;

  // One-pass f32 -> f16 weight staging in L2-resident scratch.
  cvt_f32_to_f16<<<(513 * 256 + 255) / 256, 256, 0, stream>>>(qkv_w,  w16,           513 * 256);
  cvt_f32_to_f16<<<(256 * 256 + 255) / 256, 256, 0, stream>>>(out_w,  w16 + W16_OUT, 256 * 256);
  cvt_f32_to_f16<<<(512 * 256 + 255) / 256, 256, 0, stream>>>(ffn1_w, w16 + W16_F1,  512 * 256);
  cvt_f32_to_f16<<<(256 * 512 + 255) / 256, 256, 0, stream>>>(ffn2_w, w16 + W16_F2,  256 * 512);

  // One workgroup per (b,p) slice: 4*256 = 1024 WGs, 8 waves each, ~274 KB LDS.
  linattn_ffn_fused<<<dim3(4 * PDIM), dim3(256), SMEM_BYTES, stream>>>(
      x, ln1_g, ln1_b, qkv_w, qkv_b, out_bv, ln2_g, ln2_b, ffn1_b, ffn2_b,
      w16, outp);
}
